// MultiHeadAttention_81346680586483
// MI455X (gfx1250) — matrix operand order repaired and checked by reading, not compiled, not run on hardware
//
#include <hip/hip_runtime.h>
#include <hip/hip_bf16.h>

typedef __attribute__((ext_vector_type(16))) __bf16 v16bf;
typedef __attribute__((ext_vector_type(8)))  float  v8f;

#define S_LEN 2048
#define HIDN  1024
#define NHEAD 16
#define HDIM  64
#define BATCH 2
#define MROWS (BATCH * S_LEN) /* 4096 */

// 16-bit A-matrix 16x32 K-layout (contiguous, matches B): VGPR v, lane-group g covers K 16g..16g+15
__device__ __forceinline__ int kbaseA(int v, int g) {
    return 2 * v + 16 * g;
}

// ---- wave32 16-lane butterfly reductions via DPP (no LDS traffic) ----
#define DPP8_SEL(s0, s1, s2, s3, s4, s5, s6, s7)                                   \
    ((s0) | ((s1) << 3) | ((s2) << 6) | ((s3) << 9) | ((s4) << 12) |               \
     ((s5) << 15) | ((s6) << 18) | ((s7) << 21))

template <int SEL>
__device__ __forceinline__ float dpp8_f(float x) {
    return __builtin_bit_cast(
        float, __builtin_amdgcn_mov_dpp8(__builtin_bit_cast(int, x), SEL));
}
template <int CTRL>
__device__ __forceinline__ float dpp16_f(float x) {
    return __builtin_bit_cast(
        float, __builtin_amdgcn_mov_dpp(__builtin_bit_cast(int, x), CTRL, 0xf, 0xf, true));
}

__device__ __forceinline__ float red16_max(float x) {
    x = fmaxf(x, dpp8_f<DPP8_SEL(1, 0, 3, 2, 5, 4, 7, 6)>(x));  // xor 1
    x = fmaxf(x, dpp8_f<DPP8_SEL(2, 3, 0, 1, 6, 7, 4, 5)>(x));  // xor 2
    x = fmaxf(x, dpp8_f<DPP8_SEL(4, 5, 6, 7, 0, 1, 2, 3)>(x));  // xor 4
    x = fmaxf(x, dpp16_f<0x128>(x));  // row_ror:8 == xor 8 within 16-lane row
    return x;
}
__device__ __forceinline__ float red16_sum(float x) {
    x += dpp8_f<DPP8_SEL(1, 0, 3, 2, 5, 4, 7, 6)>(x);
    x += dpp8_f<DPP8_SEL(2, 3, 0, 1, 6, 7, 4, 5)>(x);
    x += dpp8_f<DPP8_SEL(4, 5, 6, 7, 0, 1, 2, 3)>(x);
    x += dpp16_f<0x128>(x);
    return x;
}

// out[n,j] = sum_k A[n,k] * W[j,k] + bias[j]
// A: MROWS x HIDN (fp32 or bf16), W: HIDN x HIDN fp32.
// OUT_HEADS: write bf16 into [B*NH, S, HD]; else fp32 row-major.
template <bool A_BF16, bool OUT_HEADS>
__global__ __launch_bounds__(256) void gemm_wmma_kernel(
    const void* __restrict__ Ain, const float* __restrict__ W,
    const float* __restrict__ bias, void* __restrict__ Out) {
    const int lane = threadIdx.x & 31;
    const int wav  = threadIdx.x >> 5;
    const int g    = lane >> 4;
    const int nl   = lane & 15;
    const int row0 = blockIdx.x * 32 + (wav >> 2) * 16;
    const int col0 = blockIdx.y * 64 + (wav & 3) * 16;
    const float*  Af = (const float*)Ain;
    const __bf16* Ab = (const __bf16*)Ain;

    v8f acc = {0.f, 0.f, 0.f, 0.f, 0.f, 0.f, 0.f, 0.f};
    const int arow = row0 + nl;
    const int bcol = col0 + nl;

    for (int k0 = 0; k0 < HIDN; k0 += 32) {
        v16bf a, b;
#pragma unroll
        for (int v = 0; v < 8; ++v) {
            const int ka = k0 + kbaseA(v, g);
            if (A_BF16) {
                a[2 * v]     = Ab[arow * HIDN + ka];
                a[2 * v + 1] = Ab[arow * HIDN + ka + 1];
            } else {
                a[2 * v]     = (__bf16)Af[arow * HIDN + ka];
                a[2 * v + 1] = (__bf16)Af[arow * HIDN + ka + 1];
            }
            const int kb = k0 + 2 * v + 16 * g;  // B: lane-half covers K 0-15 / 16-31
            b[2 * v]     = (__bf16)W[bcol * HIDN + kb];
            b[2 * v + 1] = (__bf16)W[bcol * HIDN + kb + 1];
        }
        acc = __builtin_amdgcn_wmma_f32_16x16x32_bf16(false, a, false, b,
                                                      (short)0, acc, false, false);
    }

#pragma unroll
    for (int r = 0; r < 8; ++r) {
        const int row = row0 + r + 8 * g;   // C/D layout: M = r + 8*(lane/16)
        const int j   = bcol;               // N = lane%16
        const float val = acc[r] + bias[j];
        if (OUT_HEADS) {
            const int bidx = row >> 11;     // row / S_LEN
            const int s    = row & (S_LEN - 1);
            const int h    = j >> 6;
            const int d    = j & 63;
            ((__bf16*)Out)[(((size_t)(bidx * NHEAD + h) * S_LEN + s) << 6) + d] = (__bf16)val;
        } else {
            ((float*)Out)[(size_t)row * HIDN + j] = val;
        }
    }
}

// In-place RoPE on q,k bf16 buffers laid out [B*NH, S, HD]
__global__ __launch_bounds__(256) void rope_kernel(__bf16* __restrict__ q,
                                                   __bf16* __restrict__ k) {
    const int idx = blockIdx.x * blockDim.x + threadIdx.x;  // total = BH*S*32
    const int i   = idx & 31;
    const int s   = (idx >> 5) & (S_LEN - 1);
    const int bh  = idx >> 16;  // 32 * 2048 = 2^16 per (b,h)

    const float inv = __expf(-(float)(2 * i) * (9.2103403719762f / 64.0f)); // 10000^(-2i/64)
    const float fr  = (float)s * inv;
    float sn, cs;
    __sincosf(fr, &sn, &cs);

    const size_t base = ((size_t)bh * S_LEN + s) * HDIM;
    {
        const float a = (float)q[base + i], b = (float)q[base + i + 32];
        q[base + i]      = (__bf16)(a * cs - b * sn);
        q[base + i + 32] = (__bf16)(b * cs + a * sn);
    }
    {
        const float a = (float)k[base + i], b = (float)k[base + i + 32];
        k[base + i]      = (__bf16)(a * cs - b * sn);
        k[base + i + 32] = (__bf16)(b * cs + a * sn);
    }
}

// Flash-attention: grid (S/128, B*NH), 8 waves/block, wave owns 16 query rows.
__global__ __launch_bounds__(256) void attn_kernel(const __bf16* __restrict__ q,
                                                   const __bf16* __restrict__ k,
                                                   const __bf16* __restrict__ v,
                                                   __bf16* __restrict__ out) {
    __shared__ __bf16 pbuf[8][16][36];  // per-wave 16x32 P tile, padded stride

    const int lane  = threadIdx.x & 31;
    const int wav   = threadIdx.x >> 5;
    const int g     = lane >> 4;
    const int nl    = lane & 15;
    const int bh    = blockIdx.y;
    const int qbase = blockIdx.x * 128 + wav * 16;

    const __bf16* qh = q + (size_t)bh * S_LEN * HDIM;
    const __bf16* kh = k + (size_t)bh * S_LEN * HDIM;
    const __bf16* vh = v + (size_t)bh * S_LEN * HDIM;

    // Q fragment: 16 rows x 64 dims = two 16x32 A-fragments
    v16bf aq0, aq1;
    {
        const int row = qbase + nl;
#pragma unroll
        for (int vi = 0; vi < 8; ++vi) {
            const int kk = kbaseA(vi, g);
            aq0[2 * vi]     = qh[row * HDIM + kk];
            aq0[2 * vi + 1] = qh[row * HDIM + kk + 1];
            aq1[2 * vi]     = qh[row * HDIM + 32 + kk];
            aq1[2 * vi + 1] = qh[row * HDIM + 32 + kk + 1];
        }
    }

    v8f acc[4];
#pragma unroll
    for (int t = 0; t < 4; ++t) acc[t] = (v8f){0.f, 0.f, 0.f, 0.f, 0.f, 0.f, 0.f, 0.f};
    float mr[8], lr[8];
#pragma unroll
    for (int r = 0; r < 8; ++r) { mr[r] = -1e30f; lr[r] = 0.f; }

    for (int kt = 0; kt < S_LEN; kt += 32) {
        // prefetch next key tile of K and V into cache (global_prefetch_b8)
        if (kt + 32 < S_LEN) {
            __builtin_prefetch((const void*)(kh + (size_t)(kt + 32) * HDIM + lane * 64), 0, 0);
            __builtin_prefetch((const void*)(vh + (size_t)(kt + 32) * HDIM + lane * 64), 0, 0);
        }

        // ---- scores: two 16x16 C tiles (keys kt..+15, kt+16..+31) ----
        v16bf b00, b01, b10, b11;  // [keyhalf][dimhalf]
        const int key0 = kt + nl;
        const int key1 = kt + 16 + nl;
#pragma unroll
        for (int vi = 0; vi < 8; ++vi) {
            const int d0 = 2 * vi + 16 * g;
            const int d1 = 32 + d0;
            b00[2 * vi] = kh[key0 * HDIM + d0]; b00[2 * vi + 1] = kh[key0 * HDIM + d0 + 1];
            b01[2 * vi] = kh[key0 * HDIM + d1]; b01[2 * vi + 1] = kh[key0 * HDIM + d1 + 1];
            b10[2 * vi] = kh[key1 * HDIM + d0]; b10[2 * vi + 1] = kh[key1 * HDIM + d0 + 1];
            b11[2 * vi] = kh[key1 * HDIM + d1]; b11[2 * vi + 1] = kh[key1 * HDIM + d1 + 1];
        }
        v8f s0 = (v8f){0.f, 0.f, 0.f, 0.f, 0.f, 0.f, 0.f, 0.f};
        v8f s1 = s0;
        s0 = __builtin_amdgcn_wmma_f32_16x16x32_bf16(false, aq0, false, b00, (short)0, s0, false, false);
        s0 = __builtin_amdgcn_wmma_f32_16x16x32_bf16(false, aq1, false, b01, (short)0, s0, false, false);
        s1 = __builtin_amdgcn_wmma_f32_16x16x32_bf16(false, aq0, false, b10, (short)0, s1, false, false);
        s1 = __builtin_amdgcn_wmma_f32_16x16x32_bf16(false, aq1, false, b11, (short)0, s1, false, false);

        // ---- online softmax row update (DPP reductions) + write P tile to LDS ----
#pragma unroll
        for (int r = 0; r < 8; ++r) {
            const float x0 = s0[r] * 0.125f;  // 1/sqrt(64)
            const float x1 = s1[r] * 0.125f;
            const float mx = red16_max(fmaxf(x0, x1));
            const float nm = fmaxf(mr[r], mx);
            const float p0 = __expf(x0 - nm);
            const float p1 = __expf(x1 - nm);
            const float rs = red16_sum(p0 + p1);
            const float al = __expf(mr[r] - nm);
            lr[r] = lr[r] * al + rs;
            mr[r] = nm;
            acc[0][r] *= al; acc[1][r] *= al; acc[2][r] *= al; acc[3][r] *= al;
            pbuf[wav][r + 8 * g][nl]      = (__bf16)p0;
            pbuf[wav][r + 8 * g][16 + nl] = (__bf16)p1;
        }
        asm volatile("s_wait_dscnt 0" ::: "memory");  // CDNA5 split DS counter

        // ---- re-read P in A-fragment layout ----
        v16bf pa;
#pragma unroll
        for (int vi = 0; vi < 8; ++vi) {
            const int kk = kbaseA(vi, g);
            pa[2 * vi]     = pbuf[wav][nl][kk];
            pa[2 * vi + 1] = pbuf[wav][nl][kk + 1];
        }

        // ---- out += P (16x32) @ V (32x64) ----
#pragma unroll
        for (int t = 0; t < 4; ++t) {
            v16bf bv;
#pragma unroll
            for (int vi = 0; vi < 8; ++vi) {
                const int key = kt + 2 * vi + 16 * g;
                const int d   = t * 16 + nl;
                bv[2 * vi]     = vh[key * HDIM + d];
                bv[2 * vi + 1] = vh[(key + 1) * HDIM + d];
            }
            acc[t] = __builtin_amdgcn_wmma_f32_16x16x32_bf16(false, pa, false, bv,
                                                             (short)0, acc[t], false, false);
        }
    }

    // ---- normalize + store bf16 into [B, S, HID] ----
    const int bq = bh / NHEAD;
    const int h  = bh % NHEAD;
#pragma unroll
    for (int r = 0; r < 8; ++r) {
        const int   srow = qbase + r + 8 * g;
        const float invl = 1.0f / lr[r];
        const size_t o = ((size_t)(bq * S_LEN + srow)) * HIDN + h * HDIM + nl;
        out[o +  0] = (__bf16)(acc[0][r] * invl);
        out[o + 16] = (__bf16)(acc[1][r] * invl);
        out[o + 32] = (__bf16)(acc[2][r] * invl);
        out[o + 48] = (__bf16)(acc[3][r] * invl);
    }
}

extern "C" void kernel_launch(void* const* d_in, const int* in_sizes, int n_in,
                              void* d_out, int out_size, void* d_ws, size_t ws_size,
                              hipStream_t stream) {
    const float* x  = (const float*)d_in[0];
    const float* wq = (const float*)d_in[1];
    const float* bq = (const float*)d_in[2];
    const float* wk = (const float*)d_in[3];
    const float* bk = (const float*)d_in[4];
    const float* wv = (const float*)d_in[5];
    const float* bv = (const float*)d_in[6];
    const float* wo = (const float*)d_in[7];
    const float* bo = (const float*)d_in[8];
    float* out = (float*)d_out;

    char* ws = (char*)d_ws;
    const size_t mat = (size_t)MROWS * HIDN * sizeof(__bf16);  // 8 MB each
    __bf16* qbuf = (__bf16*)(ws);
    __bf16* kbuf = (__bf16*)(ws + mat);
    __bf16* vbuf = (__bf16*)(ws + 2 * mat);
    __bf16* abuf = (__bf16*)(ws + 3 * mat);

    const dim3 gemm_grid(MROWS / 32, HIDN / 64);  // 128 x 16
    gemm_wmma_kernel<false, true><<<gemm_grid, 256, 0, stream>>>(x, wq, bq, qbuf);
    gemm_wmma_kernel<false, true><<<gemm_grid, 256, 0, stream>>>(x, wk, bk, kbuf);
    gemm_wmma_kernel<false, true><<<gemm_grid, 256, 0, stream>>>(x, wv, bv, vbuf);

    rope_kernel<<<(BATCH * NHEAD * S_LEN * 32) / 256, 256, 0, stream>>>(qbuf, kbuf);

    attn_kernel<<<dim3(S_LEN / 128, BATCH * NHEAD), 256, 0, stream>>>(qbuf, kbuf, vbuf, abuf);

    gemm_wmma_kernel<true, false><<<gemm_grid, 256, 0, stream>>>(abuf, wo, bo, out);
}
